// SubjectLayers_33397665694369
// MI455X (gfx1250) — matrix operand order repaired and checked by reading, not compiled, not run on hardware
//
#include <hip/hip_runtime.h>

// ---------------------------------------------------------------------------
// SubjectLayers: out[b,d,t] = sum_c x[b,c,t] * w[subjects[b],c,d] + bias[s,0,d]
// B=128, C=256, T=2048, S=64, D=256  (fp32 in/out)
//
// gfx1250 strategy (round 3):
//  - bf16x3 split GEMM on v_wmma_f32_16x16x32_bf16 (near-f32 accuracy).
//  - 128(D) x 128(T) tile per 256-thread WG; 8 waves as 2(D) x 4(T), each
//    wave computes 64x32 = 4x2 WMMA accumulators -> 24 WMMA per chunk,
//    doubling matrix work per staged element vs round 2 (VALU was the limit).
//  - f32 -> bf16 hi/lo conversion once per element in registers at staging
//    (compiler emits v_cvt_pk_bf16_f32); LDS holds transposed fragment-native
//    bf16 rows so every fragment is 2x ds_load_b128, zero inner VALU.
//  - Single LDS buffer (40.5 KB, double would exceed the 64 KB static cap);
//    global loads for chunk i+1 issue before chunk i's WMMAs (register-held
//    pipeline), conversion+stores land between two cheap barriers.
// ---------------------------------------------------------------------------

typedef __attribute__((ext_vector_type(16))) __bf16        v16bf;
typedef __attribute__((ext_vector_type(8)))  float         v8f;
typedef __attribute__((ext_vector_type(4)))  unsigned int  v4u;

#define B_N 128
#define C_N 256
#define T_N 2048
#define S_N 64
#define D_N 256

#define KC      32          // K chunk (one WMMA K-depth per hi/lo pass)
#define NCHUNK  (C_N / KC)  // 8
#define TILE_T  128         // WG tile
#define TILE_D  128

// Transposed LDS rows: 32 bf16 k-values, stride 40 u16 (80B, 16B-aligned);
// rows in odd 16-blocks get +64 u16 (128B) so the half-wave K-offset lands
// on the other 32 banks -> fragment ds_load_b128 at full LDS capacity.
#define ROWU    40
#define BLKSKIP 64
#define LTOT    (127 * ROWU + BLKSKIP + ROWU)   // 5184 u16 (128 rows)

union F16 {                     // WMMA 16-bit fragment: 8 VGPRs
    v16bf v;
    v4u   q[2];
};

static __device__ __forceinline__ unsigned short bf16h(float f) {
    __bf16 h = (__bf16)f;                       // -> v_cvt_pk_bf16_f32 pairs
    return __builtin_bit_cast(unsigned short, h);
}
static __device__ __forceinline__ float bf16f(unsigned short u) {
    return __uint_as_float((unsigned int)u << 16);
}
static __device__ __forceinline__ unsigned int pack2(unsigned short a,
                                                     unsigned short b) {
    return (unsigned int)a | ((unsigned int)b << 16);
}
static __device__ __forceinline__ int rowoff(int row) {   // u16 offset of row
    return row * ROWU + (((row >> 4) & 1) << 6);
}

__global__ __launch_bounds__(256)
void subject_layers_wmma(const float* __restrict__ x,
                         const int*   __restrict__ subjects,
                         const float* __restrict__ w,
                         const float* __restrict__ bias,
                         float*       __restrict__ out)
{
    // transposed bf16 hi/lo tiles, single buffer: 4 * 5184 * 2B = 40.5 KB
    __shared__ unsigned short Xhi[LTOT];   // [t 0..127][k 0..31]
    __shared__ unsigned short Xlo[LTOT];
    __shared__ unsigned short Whi[LTOT];   // [d 0..127][k 0..31]
    __shared__ unsigned short Wlo[LTOT];

    const int tid  = threadIdx.x;
    const int b    = blockIdx.z;
    const int d0   = blockIdx.y * TILE_D;
    const int t0   = blockIdx.x * TILE_T;
    const int s    = subjects[b];

    const float* xb = x + (size_t)b * C_N * T_N + t0;
    const float* wb = w + (size_t)s * C_N * D_N + d0;

    const int wave = tid >> 5;          // 0..7 (also the staging k-group)
    const int lane = tid & 31;
    const int g    = lane >> 4;         // half-wave group
    const int ln   = lane & 15;
    const int dW   = (wave >> 2) * 64;  // wave's D offset (0/64)
    const int tW   = (wave & 3) * 32;   // wave's T offset (0..96)

    const int kg4 = wave * 4;           // this thread stages k rows kg4..kg4+3
    const int cg4 = lane * 4;           // column group (t for X, d for W)

    // staging registers for the next chunk: 4k x 4col each
    float xv[4][4];
    float wv[4][4];

    // ---- load chunk 0 ---------------------------------------------------
    #pragma unroll
    for (int r = 0; r < 4; ++r) {
        float4 v = *(const float4*)(xb + (size_t)(kg4 + r) * T_N + cg4);
        xv[r][0] = v.x; xv[r][1] = v.y; xv[r][2] = v.z; xv[r][3] = v.w;
        float4 u = *(const float4*)(wb + (size_t)(kg4 + r) * D_N + cg4);
        wv[r][0] = u.x; wv[r][1] = u.y; wv[r][2] = u.z; wv[r][3] = u.w;
    }
    // ---- convert + store chunk 0 ---------------------------------------
    #pragma unroll
    for (int c = 0; c < 4; ++c) {
        unsigned short h[4], l[4], hw[4], lw[4];
        #pragma unroll
        for (int r = 0; r < 4; ++r) {
            h[r]  = bf16h(xv[r][c]);
            l[r]  = bf16h(xv[r][c] - bf16f(h[r]));
            hw[r] = bf16h(wv[r][c]);
            lw[r] = bf16h(wv[r][c] - bf16f(hw[r]));
        }
        int base = rowoff(cg4 + c) + kg4;
        *(uint2*)(&Xhi[base]) = make_uint2(pack2(h[0],  h[1]),  pack2(h[2],  h[3]));
        *(uint2*)(&Xlo[base]) = make_uint2(pack2(l[0],  l[1]),  pack2(l[2],  l[3]));
        *(uint2*)(&Whi[base]) = make_uint2(pack2(hw[0], hw[1]), pack2(hw[2], hw[3]));
        *(uint2*)(&Wlo[base]) = make_uint2(pack2(lw[0], lw[1]), pack2(lw[2], lw[3]));
    }
    __syncthreads();

    v8f acc[4][2] = {};

    for (int kci = 0; kci < NCHUNK; ++kci) {
        const bool more = (kci + 1 < NCHUNK);

        // ---- issue next chunk's global loads (overlap with WMMAs) ------
        if (more) {
            const int kn = (kci + 1) * KC;
            #pragma unroll
            for (int r = 0; r < 4; ++r) {
                float4 v = *(const float4*)(xb + (size_t)(kn + kg4 + r) * T_N + cg4);
                xv[r][0] = v.x; xv[r][1] = v.y; xv[r][2] = v.z; xv[r][3] = v.w;
                float4 u = *(const float4*)(wb + (size_t)(kn + kg4 + r) * D_N + cg4);
                wv[r][0] = u.x; wv[r][1] = u.y; wv[r][2] = u.z; wv[r][3] = u.w;
            }
        }

        // ---- B fragments (X'): e -> K = 16*(e>>3) + 8g + (e&7) ---------
        F16 bhi[2], blo[2];
        #pragma unroll
        for (int nt = 0; nt < 2; ++nt) {
            int base = rowoff(tW + nt * 16 + ln);
            bhi[nt].q[0] = *(const v4u*)(Xhi + base + 8 * g);
            bhi[nt].q[1] = *(const v4u*)(Xhi + base + 16 + 8 * g);
            blo[nt].q[0] = *(const v4u*)(Xlo + base + 8 * g);
            blo[nt].q[1] = *(const v4u*)(Xlo + base + 16 + 8 * g);
        }
        // ---- A fragments (W') per mt, then bf16x3 WMMAs ----------------
        #pragma unroll
        for (int mt = 0; mt < 4; ++mt) {
            F16 ahi, alo;                // e -> K = 16*(e>>3) + 8g + (e&7)
            int base = rowoff(dW + mt * 16 + ln);
            ahi.q[0] = *(const v4u*)(Whi + base + 8 * g);
            ahi.q[1] = *(const v4u*)(Whi + base + 16 + 8 * g);
            alo.q[0] = *(const v4u*)(Wlo + base + 8 * g);
            alo.q[1] = *(const v4u*)(Wlo + base + 16 + 8 * g);
            #pragma unroll
            for (int nt = 0; nt < 2; ++nt) {
                acc[mt][nt] = __builtin_amdgcn_wmma_f32_16x16x32_bf16(
                    false, alo.v, false, bhi[nt].v, (short)0, acc[mt][nt],
                    false, false);
                acc[mt][nt] = __builtin_amdgcn_wmma_f32_16x16x32_bf16(
                    false, ahi.v, false, blo[nt].v, (short)0, acc[mt][nt],
                    false, false);
                acc[mt][nt] = __builtin_amdgcn_wmma_f32_16x16x32_bf16(
                    false, ahi.v, false, bhi[nt].v, (short)0, acc[mt][nt],
                    false, false);
            }
        }
        __syncthreads();                 // all waves done reading the buffer

        // ---- convert + store next chunk into the (single) buffer -------
        if (more) {
            #pragma unroll
            for (int c = 0; c < 4; ++c) {
                unsigned short h[4], l[4], hw[4], lw[4];
                #pragma unroll
                for (int r = 0; r < 4; ++r) {
                    h[r]  = bf16h(xv[r][c]);
                    l[r]  = bf16h(xv[r][c] - bf16f(h[r]));
                    hw[r] = bf16h(wv[r][c]);
                    lw[r] = bf16h(wv[r][c] - bf16f(hw[r]));
                }
                int base = rowoff(cg4 + c) + kg4;
                *(uint2*)(&Xhi[base]) = make_uint2(pack2(h[0],  h[1]),  pack2(h[2],  h[3]));
                *(uint2*)(&Xlo[base]) = make_uint2(pack2(l[0],  l[1]),  pack2(l[2],  l[3]));
                *(uint2*)(&Whi[base]) = make_uint2(pack2(hw[0], hw[1]), pack2(hw[2], hw[3]));
                *(uint2*)(&Wlo[base]) = make_uint2(pack2(lw[0], lw[1]), pack2(lw[2], lw[3]));
            }
            __syncthreads();             // stores visible before next reads
        }
    }

    // ---- epilogue: bias + store (C/D layout: VGPR r -> M = r + 8g, N = ln)
    float* ob = out + (size_t)b * D_N * T_N;
    #pragma unroll
    for (int mt = 0; mt < 4; ++mt) {
        #pragma unroll
        for (int r = 0; r < 8; ++r) {
            int   d  = d0 + dW + mt * 16 + r + (g << 3);
            float bv = bias[s * D_N + d];
            #pragma unroll
            for (int nt = 0; nt < 2; ++nt) {
                int t = t0 + tW + nt * 16 + ln;
                ob[(size_t)d * T_N + t] = acc[mt][nt][r] + bv;
            }
        }
    }
}

extern "C" void kernel_launch(void* const* d_in, const int* in_sizes, int n_in,
                              void* d_out, int out_size, void* d_ws, size_t ws_size,
                              hipStream_t stream) {
    const float* x        = (const float*)d_in[0];   // (B, C, T) fp32
    const int*   subjects = (const int*)  d_in[1];   // (B,) int32
    const float* weights  = (const float*)d_in[2];   // (S, C, D) fp32
    const float* biases   = (const float*)d_in[3];   // (S, 1, D) fp32
    float*       out      = (float*)d_out;           // (B, D, T) fp32

    dim3 grid(T_N / TILE_T, D_N / TILE_D, B_N);      // 16 x 2 x 128
    dim3 block(256);
    subject_layers_wmma<<<grid, block, 0, stream>>>(x, subjects, weights, biases, out);
}
